// CausalSelfAttention_1700807050051
// MI455X (gfx1250) — compile-verified, matrix operand order checked by reading, not run on hardware
//
#include <hip/hip_runtime.h>
#include <hip/hip_bf16.h>

typedef __attribute__((ext_vector_type(16))) _Float16 v16h;
typedef __attribute__((ext_vector_type(8)))  _Float16 v8h;
typedef __attribute__((ext_vector_type(8)))  float    v8f;
typedef __attribute__((ext_vector_type(4)))  int      v4i;

#define B_NUM   4
#define T_SEQ   2048
#define C_DIM   1024
#define H_NUM   16
#define D_HEAD  64
#define M_ROWS  8192        // B*T
#define Y_ELEMS 8388608     // B*T*C
#define KV_ELEMS 8388608    // B*H*T*D

#if __has_builtin(__builtin_amdgcn_global_load_async_to_lds_b128)
#define HAVE_ASYNC_LDS 1
#else
#define HAVE_ASYNC_LDS 0
#endif

// Builtin signature (from hipcc diagnostic): (v4i AS1*, v4i AS3*, imm, imm)
#define AS1_V4I(p) ((__attribute__((address_space(1))) v4i*)(p))
#define AS3_V4I(p) ((__attribute__((address_space(3))) v4i*)(p))

static __device__ __forceinline__ void wait_async_zero() {
#if __has_builtin(__builtin_amdgcn_s_wait_asynccnt)
  __builtin_amdgcn_s_wait_asynccnt(0);
#else
  asm volatile("s_wait_asynccnt 0x0" ::: "memory");
#endif
}

// ---------------------------------------------------------------------------
// Fragment load: 16x32 tile (A) or 32x16-as-K-contiguous (B) of f16.
// CDNA5 16-bit A-matrix layout (wave32):
//   lanes 0-15 : row = lane,    halves 0-7 = K+0..7,  halves 8-15 = K+16..23
//   lanes 16-31: row = lane-16, halves 0-7 = K+8..15, halves 8-15 = K+24..31
// B fragment is symmetric with "row" = output column, K contiguous in memory.
// ---------------------------------------------------------------------------
static __device__ __forceinline__ v16h load_frag_k32(const _Float16* base, int ld,
                                                     int rc0, int k0, int lane) {
  const int grp = (lane >> 4) & 1;
  const int rr  = lane & 15;
  const _Float16* p = base + (size_t)(rc0 + rr) * ld + k0 + grp * 8;
  v8h lo = *(const v8h*)(p);        // K = k0 + grp*8 + 0..7
  v8h hi = *(const v8h*)(p + 16);   // K = k0 + 16 + grp*8 + 0..7
  v16h a;
#pragma unroll
  for (int i = 0; i < 8; ++i) { a[i] = lo[i]; a[i + 8] = hi[i]; }
  return a;
}

// A-fragment of P (16x32 f16) from LDS (row-major, ld = 32)
static __device__ __forceinline__ v16h load_afrag_ldsP(const _Float16* P, int lane) {
  const int grp = (lane >> 4) & 1;
  const int rr  = lane & 15;
  v16h a;
#pragma unroll
  for (int i = 0; i < 8; ++i) {
    a[i]     = P[rr * 32 + grp * 8 + i];
    a[i + 8] = P[rr * 32 + 16 + grp * 8 + i];
  }
  return a;
}

// B-fragment (32 K-rows x 16 cols) of V from LDS (row-major 32x64, ld = 64)
static __device__ __forceinline__ v16h load_bfrag_ldsV(const _Float16* V, int n0, int lane) {
  const int grp = (lane >> 4) & 1;
  const int cc  = lane & 15;
  v16h b;
#pragma unroll
  for (int i = 0; i < 8; ++i) {
    b[i]     = V[(grp * 8 + i) * D_HEAD + n0 + cc];
    b[i + 8] = V[(16 + grp * 8 + i) * D_HEAD + n0 + cc];
  }
  return b;
}

// ---------------------------------------------------------------------------
// Elementwise fp32 -> f16
// ---------------------------------------------------------------------------
__global__ void cvt_f32_to_f16_kernel(const float* __restrict__ in,
                                      _Float16* __restrict__ out, int n) {
  int i = blockIdx.x * blockDim.x + threadIdx.x;
  if (i < n) out[i] = (_Float16)in[i];
}

// Transpose + convert weight: out[n*C + k] = (f16) in[k*C + n]   (C = 1024)
__global__ void cvt_transpose_w_kernel(const float* __restrict__ in,
                                       _Float16* __restrict__ out) {
  int i  = blockIdx.x * blockDim.x + threadIdx.x;   // over C*C
  int nn = i >> 10;
  int kk = i & 1023;
  out[i] = (_Float16)in[(size_t)kk * C_DIM + nn];
}

// ---------------------------------------------------------------------------
// QKV projection: one wave computes a 32x64 tile of (x @ w + b)
// via 2 A-fragments x 4 B-fragments -> 8 WMMA accumulators (register blocked).
// grid = (M/32, C/64, 3)  z: 0=Q, 1=K, 2=V
// Writes f16 Q/K/V in (B,H,T,D) layout; K,V also as fp32 into `present`.
// ---------------------------------------------------------------------------
__global__ __launch_bounds__(32) void qkv_gemm_kernel(
    const _Float16* __restrict__ xh,
    const _Float16* __restrict__ wqT, const _Float16* __restrict__ wkT,
    const _Float16* __restrict__ wvT,
    const float* __restrict__ bq, const float* __restrict__ bk,
    const float* __restrict__ bv,
    _Float16* __restrict__ qh, _Float16* __restrict__ kh,
    _Float16* __restrict__ vh,
    float* __restrict__ present) {
  const int lane  = threadIdx.x;
  const int bm    = blockIdx.x;
  const int bn    = blockIdx.y;
  const int which = blockIdx.z;
  const _Float16* wT   = (which == 0) ? wqT : ((which == 1) ? wkT : wvT);
  const float*    bias = (which == 0) ? bq  : ((which == 1) ? bk  : bv);

  v8f acc[2][4];
#pragma unroll
  for (int i = 0; i < 2; ++i)
#pragma unroll
    for (int j = 0; j < 4; ++j) acc[i][j] = (v8f){};

  for (int k0 = 0; k0 < C_DIM; k0 += 32) {
    if (k0 + 32 < C_DIM) {
      __builtin_prefetch(xh + (size_t)(bm * 32 + (lane & 15)) * C_DIM + k0 + 32, 0, 1);
      __builtin_prefetch(xh + (size_t)(bm * 32 + 16 + (lane & 15)) * C_DIM + k0 + 32, 0, 1);
    }
    v16h a0 = load_frag_k32(xh, C_DIM, bm * 32,      k0, lane);
    v16h a1 = load_frag_k32(xh, C_DIM, bm * 32 + 16, k0, lane);
    v16h b0 = load_frag_k32(wT, C_DIM, bn * 64,      k0, lane);
    v16h b1 = load_frag_k32(wT, C_DIM, bn * 64 + 16, k0, lane);
    v16h b2 = load_frag_k32(wT, C_DIM, bn * 64 + 32, k0, lane);
    v16h b3 = load_frag_k32(wT, C_DIM, bn * 64 + 48, k0, lane);
    acc[0][0] = __builtin_amdgcn_wmma_f32_16x16x32_f16(false, a0, false, b0, (short)0, acc[0][0], false, false);
    acc[0][1] = __builtin_amdgcn_wmma_f32_16x16x32_f16(false, a0, false, b1, (short)0, acc[0][1], false, false);
    acc[0][2] = __builtin_amdgcn_wmma_f32_16x16x32_f16(false, a0, false, b2, (short)0, acc[0][2], false, false);
    acc[0][3] = __builtin_amdgcn_wmma_f32_16x16x32_f16(false, a0, false, b3, (short)0, acc[0][3], false, false);
    acc[1][0] = __builtin_amdgcn_wmma_f32_16x16x32_f16(false, a1, false, b0, (short)0, acc[1][0], false, false);
    acc[1][1] = __builtin_amdgcn_wmma_f32_16x16x32_f16(false, a1, false, b1, (short)0, acc[1][1], false, false);
    acc[1][2] = __builtin_amdgcn_wmma_f32_16x16x32_f16(false, a1, false, b2, (short)0, acc[1][2], false, false);
    acc[1][3] = __builtin_amdgcn_wmma_f32_16x16x32_f16(false, a1, false, b3, (short)0, acc[1][3], false, false);
  }

  const int grp = lane >> 4;
  const int col = lane & 15;
#pragma unroll
  for (int j = 0; j < 4; ++j) {
    const int n  = bn * 64 + j * 16 + col;
    const float bb = bias[n];
    const int h = n >> 6;        // n / D_HEAD
    const int d = n & 63;        // n % D_HEAD
#pragma unroll
    for (int i = 0; i < 2; ++i) {
#pragma unroll
      for (int r = 0; r < 8; ++r) {
        const int m  = bm * 32 + i * 16 + r + grp * 8;  // global row = b*T + t
        const int b_ = m >> 11;                         // / T_SEQ
        const int t  = m & 2047;                        // % T_SEQ
        const size_t idx = ((size_t)(b_ * H_NUM + h) * T_SEQ + t) * D_HEAD + d;
        const float val = acc[i][j][r] + bb;
        if (which == 0) {
          qh[idx] = (_Float16)val;
        } else if (which == 1) {
          kh[idx] = (_Float16)val;
          present[idx] = val;                   // present[0] = K
        } else {
          vh[idx] = (_Float16)val;
          present[KV_ELEMS + idx] = val;        // present[1] = V
        }
      }
    }
  }
}

// ---------------------------------------------------------------------------
// Flash attention: one wave handles 16 query rows of one (b,h).
// grid = (T/16, B*H)
// ---------------------------------------------------------------------------
__global__ __launch_bounds__(32) void attn_flash_kernel(
    const _Float16* __restrict__ qh, const _Float16* __restrict__ kh,
    const _Float16* __restrict__ vh, _Float16* __restrict__ oh) {
  __shared__ __align__(16) _Float16 Pt[16 * 32];
  __shared__ __align__(16) _Float16 Vt[32 * D_HEAD];

  const int lane = threadIdx.x;
  const int grp  = lane >> 4;
  const int col  = lane & 15;
  const int q0   = blockIdx.x * 16;
  const int bh   = blockIdx.y;                // b*H + h
  const _Float16* qb = qh + (size_t)bh * T_SEQ * D_HEAD;
  const _Float16* kb = kh + (size_t)bh * T_SEQ * D_HEAD;
  const _Float16* vb = vh + (size_t)bh * T_SEQ * D_HEAD;

  const v16h qa0 = load_frag_k32(qb, D_HEAD, q0, 0, lane);
  const v16h qa1 = load_frag_k32(qb, D_HEAD, q0, 32, lane);

  float mrow[8], lrow[8];
  v8f o0 = {}, o1 = {}, o2 = {}, o3 = {};
#pragma unroll
  for (int r = 0; r < 8; ++r) { mrow[r] = -__builtin_inff(); lrow[r] = 0.0f; }

  const int kcmax = (q0 + 15) >> 5;           // last 32-key chunk (causal)
  for (int kc = 0; kc <= kcmax; ++kc) {
    const int k0 = kc * 32;

    // --- stage V chunk (32 x 64 f16 = 4096 contiguous bytes) into LDS ---
#if HAVE_ASYNC_LDS
    {
      // CDNA5 async copy: GLOBAL_LOAD_ASYNC_TO_LDS_B128, tracked by ASYNCcnt.
      const char* gsrc = (const char*)(vb + (size_t)k0 * D_HEAD);
      char* ldst = (char*)&Vt[0];
#pragma unroll
      for (int i = 0; i < 8; ++i) {
        __builtin_amdgcn_global_load_async_to_lds_b128(
            AS1_V4I(gsrc + lane * 16 + i * 512),
            AS3_V4I(ldst + lane * 16 + i * 512), 0, 0);
      }
    }
#else
    {
      const _Float16* vp = vb + (size_t)(k0 + lane) * D_HEAD;
#pragma unroll
      for (int i = 0; i < 8; ++i)
        *(v8h*)&Vt[lane * D_HEAD + i * 8] = *(const v8h*)(vp + i * 8);
    }
#endif

    // --- S = Q K^T for a 16x32 score tile (two 16x16 fragments) ---
    v8f s0 = {}, s1 = {};
    {
      v16h b00 = load_frag_k32(kb, D_HEAD, k0,       0,  lane);
      v16h b01 = load_frag_k32(kb, D_HEAD, k0,       32, lane);
      v16h b10 = load_frag_k32(kb, D_HEAD, k0 + 16,  0,  lane);
      v16h b11 = load_frag_k32(kb, D_HEAD, k0 + 16,  32, lane);
      s0 = __builtin_amdgcn_wmma_f32_16x16x32_f16(false, qa0, false, b00, (short)0, s0, false, false);
      s0 = __builtin_amdgcn_wmma_f32_16x16x32_f16(false, qa1, false, b01, (short)0, s0, false, false);
      s1 = __builtin_amdgcn_wmma_f32_16x16x32_f16(false, qa0, false, b10, (short)0, s1, false, false);
      s1 = __builtin_amdgcn_wmma_f32_16x16x32_f16(false, qa1, false, b11, (short)0, s1, false, false);
    }

    // --- online softmax (rows of the C fragment live in 16-lane groups) ---
    const float sc = 0.125f;                  // 1/sqrt(64)
#pragma unroll
    for (int r = 0; r < 8; ++r) {
      const int trow = q0 + r + grp * 8;
      float a0 = s0[r] * sc;
      float a1 = s1[r] * sc;
      a0 = (k0 + col      > trow) ? -__builtin_inff() : a0;  // causal mask
      a1 = (k0 + 16 + col > trow) ? -__builtin_inff() : a1;
      float mx = fmaxf(a0, a1);
      mx = fmaxf(mx, __shfl_xor(mx, 1, 32));
      mx = fmaxf(mx, __shfl_xor(mx, 2, 32));
      mx = fmaxf(mx, __shfl_xor(mx, 4, 32));
      mx = fmaxf(mx, __shfl_xor(mx, 8, 32));
      const float mnew  = fmaxf(mrow[r], mx);
      const float alpha = __expf(mrow[r] - mnew);
      const float e0 = __expf(a0 - mnew);
      const float e1 = __expf(a1 - mnew);
      float rs = e0 + e1;
      rs += __shfl_xor(rs, 1, 32);
      rs += __shfl_xor(rs, 2, 32);
      rs += __shfl_xor(rs, 4, 32);
      rs += __shfl_xor(rs, 8, 32);
      lrow[r] = lrow[r] * alpha + rs;
      mrow[r] = mnew;
      o0[r] *= alpha; o1[r] *= alpha; o2[r] *= alpha; o3[r] *= alpha;
      Pt[(r + grp * 8) * 32 + col]      = (_Float16)e0;
      Pt[(r + grp * 8) * 32 + 16 + col] = (_Float16)e1;
    }
#if HAVE_ASYNC_LDS
    wait_async_zero();                        // V tile resident in LDS
#endif
    __syncthreads();

    // --- O += P V  (16x32 @ 32x64 as four 16x16 output fragments) ---
    const v16h pa  = load_afrag_ldsP(Pt, lane);
    const v16h vf0 = load_bfrag_ldsV(Vt, 0,  lane);
    const v16h vf1 = load_bfrag_ldsV(Vt, 16, lane);
    const v16h vf2 = load_bfrag_ldsV(Vt, 32, lane);
    const v16h vf3 = load_bfrag_ldsV(Vt, 48, lane);
    o0 = __builtin_amdgcn_wmma_f32_16x16x32_f16(false, pa, false, vf0, (short)0, o0, false, false);
    o1 = __builtin_amdgcn_wmma_f32_16x16x32_f16(false, pa, false, vf1, (short)0, o1, false, false);
    o2 = __builtin_amdgcn_wmma_f32_16x16x32_f16(false, pa, false, vf2, (short)0, o2, false, false);
    o3 = __builtin_amdgcn_wmma_f32_16x16x32_f16(false, pa, false, vf3, (short)0, o3, false, false);
    __syncthreads();
  }

  // --- normalize and write O in (B,T,C) f16 layout for the final GEMM ---
  const int b_ = bh >> 4;       // / H_NUM
  const int h  = bh & 15;       // % H_NUM
#pragma unroll
  for (int r = 0; r < 8; ++r) {
    const int t = q0 + r + grp * 8;
    const float inv = 1.0f / lrow[r];
    _Float16* op = oh + ((size_t)b_ * T_SEQ + t) * C_DIM + h * D_HEAD;
    op[col]      = (_Float16)(o0[r] * inv);
    op[16 + col] = (_Float16)(o1[r] * inv);
    op[32 + col] = (_Float16)(o2[r] * inv);
    op[48 + col] = (_Float16)(o3[r] * inv);
  }
}

// ---------------------------------------------------------------------------
// Output projection: y = O @ wp + bp (fp32 out), 32x64 tile per wave.
// grid = (M/32, C/64)
// ---------------------------------------------------------------------------
__global__ __launch_bounds__(32) void proj_gemm_kernel(
    const _Float16* __restrict__ ohalf, const _Float16* __restrict__ wpT,
    const float* __restrict__ bp, float* __restrict__ y) {
  const int lane = threadIdx.x;
  const int bm = blockIdx.x;
  const int bn = blockIdx.y;

  v8f acc[2][4];
#pragma unroll
  for (int i = 0; i < 2; ++i)
#pragma unroll
    for (int j = 0; j < 4; ++j) acc[i][j] = (v8f){};

  for (int k0 = 0; k0 < C_DIM; k0 += 32) {
    if (k0 + 32 < C_DIM) {
      __builtin_prefetch(ohalf + (size_t)(bm * 32 + (lane & 15)) * C_DIM + k0 + 32, 0, 1);
    }
    v16h a0 = load_frag_k32(ohalf, C_DIM, bm * 32,      k0, lane);
    v16h a1 = load_frag_k32(ohalf, C_DIM, bm * 32 + 16, k0, lane);
    v16h b0 = load_frag_k32(wpT,   C_DIM, bn * 64,      k0, lane);
    v16h b1 = load_frag_k32(wpT,   C_DIM, bn * 64 + 16, k0, lane);
    v16h b2 = load_frag_k32(wpT,   C_DIM, bn * 64 + 32, k0, lane);
    v16h b3 = load_frag_k32(wpT,   C_DIM, bn * 64 + 48, k0, lane);
    acc[0][0] = __builtin_amdgcn_wmma_f32_16x16x32_f16(false, a0, false, b0, (short)0, acc[0][0], false, false);
    acc[0][1] = __builtin_amdgcn_wmma_f32_16x16x32_f16(false, a0, false, b1, (short)0, acc[0][1], false, false);
    acc[0][2] = __builtin_amdgcn_wmma_f32_16x16x32_f16(false, a0, false, b2, (short)0, acc[0][2], false, false);
    acc[0][3] = __builtin_amdgcn_wmma_f32_16x16x32_f16(false, a0, false, b3, (short)0, acc[0][3], false, false);
    acc[1][0] = __builtin_amdgcn_wmma_f32_16x16x32_f16(false, a1, false, b0, (short)0, acc[1][0], false, false);
    acc[1][1] = __builtin_amdgcn_wmma_f32_16x16x32_f16(false, a1, false, b1, (short)0, acc[1][1], false, false);
    acc[1][2] = __builtin_amdgcn_wmma_f32_16x16x32_f16(false, a1, false, b2, (short)0, acc[1][2], false, false);
    acc[1][3] = __builtin_amdgcn_wmma_f32_16x16x32_f16(false, a1, false, b3, (short)0, acc[1][3], false, false);
  }

  const int grp = lane >> 4;
  const int col = lane & 15;
#pragma unroll
  for (int j = 0; j < 4; ++j) {
    const int n = bn * 64 + j * 16 + col;
    const float bb = bp[n];
#pragma unroll
    for (int i = 0; i < 2; ++i) {
#pragma unroll
      for (int r = 0; r < 8; ++r) {
        const int m = bm * 32 + i * 16 + r + grp * 8;
        y[(size_t)m * C_DIM + n] = acc[i][j][r] + bb;
      }
    }
  }
}

// ---------------------------------------------------------------------------
extern "C" void kernel_launch(void* const* d_in, const int* in_sizes, int n_in,
                              void* d_out, int out_size, void* d_ws, size_t ws_size,
                              hipStream_t stream) {
  const float* x  = (const float*)d_in[0];
  const float* wq = (const float*)d_in[1];
  const float* bq = (const float*)d_in[2];
  const float* wk = (const float*)d_in[3];
  const float* bk = (const float*)d_in[4];
  const float* wv = (const float*)d_in[5];
  const float* bv = (const float*)d_in[6];
  const float* wp = (const float*)d_in[7];
  const float* bp = (const float*)d_in[8];

  float* y       = (float*)d_out;
  float* present = y + Y_ELEMS;          // (2, B, H, T, D)

  // Workspace carve-up (f16 elements): ~88 MB total
  _Float16* xh  = (_Float16*)d_ws;               // B*T*C
  _Float16* wqT = xh  + Y_ELEMS;                 // C*C each
  _Float16* wkT = wqT + C_DIM * C_DIM;
  _Float16* wvT = wkT + C_DIM * C_DIM;
  _Float16* wpT = wvT + C_DIM * C_DIM;
  _Float16* qh  = wpT + C_DIM * C_DIM;           // (B,H,T,D)
  _Float16* kh  = qh  + KV_ELEMS;
  _Float16* vh  = kh  + KV_ELEMS;
  _Float16* ohh = vh  + KV_ELEMS;                // (B,T,C)

  // 1) precision conversion + weight transposes
  cvt_f32_to_f16_kernel<<<Y_ELEMS / 256, 256, 0, stream>>>(x, xh, Y_ELEMS);
  cvt_transpose_w_kernel<<<(C_DIM * C_DIM) / 256, 256, 0, stream>>>(wq, wqT);
  cvt_transpose_w_kernel<<<(C_DIM * C_DIM) / 256, 256, 0, stream>>>(wk, wkT);
  cvt_transpose_w_kernel<<<(C_DIM * C_DIM) / 256, 256, 0, stream>>>(wv, wvT);
  cvt_transpose_w_kernel<<<(C_DIM * C_DIM) / 256, 256, 0, stream>>>(wp, wpT);

  // 2) QKV projections (WMMA, 32x64 register-blocked) — also emits `present`
  dim3 gq(M_ROWS / 32, C_DIM / 64, 3);
  qkv_gemm_kernel<<<gq, 32, 0, stream>>>(xh, wqT, wkT, wvT, bq, bk, bv,
                                         qh, kh, vh, present);

  // 3) causal flash attention (WMMA + async V staging)
  dim3 ga(T_SEQ / 16, B_NUM * H_NUM);
  attn_flash_kernel<<<ga, 32, 0, stream>>>(qh, kh, vh, ohh);

  // 4) output projection (WMMA, 32x64 register-blocked)
  dim3 gp(M_ROWS / 32, C_DIM / 64);
  proj_gemm_kernel<<<gp, 32, 0, stream>>>(ohh, wpT, bp, y);
}
